// GRUPlant_36112085025198
// MI455X (gfx1250) — compile-verified, matrix operand order checked by reading, not run on hardware
//
#include <hip/hip_runtime.h>
#include <hip/hip_bf16.h>

// ---------------------------------------------------------------------------
// GRU encoder/decoder plant model for MI455X (gfx1250), wave32 + WMMA bf16.
// 16 workgroups x 16-row batch tiles; each runs the full 900-step recurrence.
// Per-step GEMMs: v_wmma_f32_16x16x32_bf16, M=16, N=1536 (96 tiles over 16
// waves). K loops are stationary 2-wide software pipelines (two fixed (a,b)
// register sets reloaded in place -> no rotation moves, 2 loads in flight).
// sched_barrier(0) fences at phase AND tile boundaries stop load hoisting
// into high-pressure regions (-> no scratch spills).
// ---------------------------------------------------------------------------

typedef __attribute__((ext_vector_type(16))) __bf16 v16bf;
typedef __attribute__((ext_vector_type(8)))  __bf16 v8bf;
typedef __attribute__((ext_vector_type(8)))  float  v8f;

#define B_    256
#define TE_   300
#define T_    600
#define NIN_  128
#define NPV_  5
#define H_    512
#define EMB_  32
#define NHE_  16
#define G3_   1536      // 3*H
#define TPB_  512       // 16 waves
#define BTILE_ 16
#define NBLK_ (B_/BTILE_)
#define XPAD_ 192       // encoder layer-0 K (160) padded to even KT (6*32)

__device__ __forceinline__ void sched_fence(){ __builtin_amdgcn_sched_barrier(0); }

__device__ __forceinline__ __bf16 f2bf(float f){
  union { float f; unsigned u; } x; x.f = f;
  unsigned r = x.u + 0x7FFFu + ((x.u >> 16) & 1u);   // round-to-nearest-even
  unsigned short h = (unsigned short)(r >> 16);
  __bf16 b; __builtin_memcpy(&b, &h, 2);
  return b;
}
__device__ __forceinline__ float sigm(float x){ return 1.f / (1.f + __expf(-x)); }

// A-fragment (16x32 bf16 tile of the 16-row activation matrix) from LDS.
// ISA layout: lanes 0-15 -> M=lane, K={0..7,16..23}; lanes 16-31 -> M=lane-16,
// K={8..15,24..31}. Two contiguous 16B halves per lane.
__device__ __forceinline__ v16bf load_a(const __bf16* A, int pitch, int kt, int lane){
  int m  = lane & 15;
  int kb = (kt << 5) + ((lane & 16) ? 8 : 0);
  const __bf16* row = A + m * pitch + kb;
  v8bf lo = *(const v8bf*)(row);
  v8bf hi = *(const v8bf*)(row + 16);
  return __builtin_shufflevector(lo, hi, 0,1,2,3,4,5,6,7,8,9,10,11,12,13,14,15);
}

// B-fragment from pre-packed global weights: fragment (nt,kt) is 512 bf16
// stored lane-major (lane*16 contiguous) -> each lane: one 32B contiguous read.
__device__ __forceinline__ v16bf load_b(const __bf16* P, int nt, int kt, int KT, int lane){
  return *(const v16bf*)(P + (((size_t)(nt * KT + kt)) << 9) + (lane << 4));
}

__device__ __forceinline__ v8f wmma_bf16(v16bf a, v16bf b, v8f c){
  return __builtin_amdgcn_wmma_f32_16x16x32_bf16(false, a, false, b, (short)0, c, false, false);
}

// Stationary 2-wide pipelined K loop: two fixed (a,b) register sets, each
// reloaded right after its WMMA consumes it -> ~2 K-chunks of loads in flight,
// no register rotation moves, no dynamic indexing. Requires even KT >= 4.
template<int KT>
__device__ __forceinline__ v8f gemm_k(const __bf16* A, int apitch,
                                      const __bf16* W, int nt, int lane, v8f acc)
{
  static_assert(KT >= 4 && (KT % 2) == 0, "");
  v16bf a0 = load_a(A, apitch, 0, lane), b0 = load_b(W, nt, 0, KT, lane);
  v16bf a1 = load_a(A, apitch, 1, lane), b1 = load_b(W, nt, 1, KT, lane);
#pragma unroll 1
  for (int kt = 0; kt < KT - 2; kt += 2){
    acc = wmma_bf16(a0, b0, acc);
    a0 = load_a(A, apitch, kt + 2, lane);
    b0 = load_b(W, nt, kt + 2, KT, lane);
    acc = wmma_bf16(a1, b1, acc);
    a1 = load_a(A, apitch, kt + 3, lane);
    b1 = load_b(W, nt, kt + 3, KT, lane);
  }
  acc = wmma_bf16(a0, b0, acc);
  return wmma_bf16(a1, b1, acc);
}

// Same, dual accumulator (n-gate tiles: full sum + h-only part share A/B).
template<int KT>
__device__ __forceinline__ void gemm_k2(const __bf16* A, int apitch,
                                        const __bf16* W, int nt, int lane,
                                        v8f& accA, v8f& accH)
{
  static_assert(KT >= 4 && (KT % 2) == 0, "");
  v16bf a0 = load_a(A, apitch, 0, lane), b0 = load_b(W, nt, 0, KT, lane);
  v16bf a1 = load_a(A, apitch, 1, lane), b1 = load_b(W, nt, 1, KT, lane);
#pragma unroll 1
  for (int kt = 0; kt < KT - 2; kt += 2){
    accA = wmma_bf16(a0, b0, accA);
    accH = wmma_bf16(a0, b0, accH);
    a0 = load_a(A, apitch, kt + 2, lane);
    b0 = load_b(W, nt, kt + 2, KT, lane);
    accA = wmma_bf16(a1, b1, accA);
    accH = wmma_bf16(a1, b1, accH);
    a1 = load_a(A, apitch, kt + 3, lane);
    b1 = load_b(W, nt, kt + 3, KT, lane);
  }
  accA = wmma_bf16(a0, b0, accA);
  accH = wmma_bf16(a0, b0, accH);
  accA = wmma_bf16(a1, b1, accA);
  accH = wmma_bf16(a1, b1, accH);
}

// gsum[16][1536] = bih + bhh + x(16xKx)@Wx^T + h(16x512)@Wh^T
// hnn [16][512]  = bhh_n + h @ Whh_n^T   (n-gate tiles 64..95 only)
// Tile map: nt = ti*16 + wave  ->  ti 0..3 are r/z tiles, ti 4..5 are n tiles.
template<int KXT>
__device__ __forceinline__ void gates_gemm(
    const __bf16* xA, int xpitch, const __bf16* Wx,
    const __bf16* hA, const __bf16* Wh,
    const float* __restrict__ bih, const float* __restrict__ bhh,
    float* gsum, float* hnn, int wave, int lane)
{
  const int ncol = lane & 15;
  const int mb   = (lane & 16) ? 8 : 0;

  // ---- r/z gate tiles (single accumulator) ----
  for (int ti = 0; ti < 4; ++ti){
    const int nt = ti * 16 + wave;
    sched_fence();                         // keep tile prologue local
    v8f accA = {};
    accA = gemm_k<KXT>(xA, xpitch, Wx, nt, lane, accA);
    accA = gemm_k<16>(hA, H_, Wh, nt, lane, accA);
    const int n = (nt << 4) + ncol;
    const float bb = bih[n] + bhh[n];
#pragma unroll
    for (int i = 0; i < 8; ++i)
      gsum[(mb + i) * G3_ + n] = accA[i] + bb;
    sched_fence();                         // keep tile epilogue local
  }

  // ---- n gate tiles (dual accumulator: full sum + h-only part) ----
  for (int ti = 0; ti < 2; ++ti){
    const int nt = 64 + ti * 16 + wave;
    sched_fence();
    v8f accA = {};
    v8f accH = {};
    accA = gemm_k<KXT>(xA, xpitch, Wx, nt, lane, accA);
    gemm_k2<16>(hA, H_, Wh, nt, lane, accA, accH);
    const int n = (nt << 4) + ncol;
    const float bi = bih[n], bh = bhh[n];
#pragma unroll
    for (int i = 0; i < 8; ++i){
      gsum[(mb + i) * G3_ + n] = accA[i] + bi + bh;
      hnn[(mb + i) * H_ + (n - 2 * H_)] = accH[i] + bh;
    }
    sched_fence();
  }
}

// h = (1-z)*tanh(g_n + (r-1)*h_n) + z*h   (PyTorch r,z,n ordering)
__device__ __forceinline__ void gru_update(const float* gsum, const float* hnn,
                                           __bf16* h, int tid)
{
  for (int idx = tid; idx < 16 * H_; idx += TPB_){
    int m = idx >> 9, j = idx & (H_ - 1);
    float r  = sigm(gsum[m * G3_ + j]);
    float z  = sigm(gsum[m * G3_ + H_ + j]);
    float g  = gsum[m * G3_ + 2 * H_ + j];
    float hn = hnn[idx];
    float nn = tanhf(g + (r - 1.f) * hn);
    float ho = (float)h[idx];
    h[idx] = f2bf((1.f - z) * nn + z * ho);
  }
}

// Pack fp32 weight W[Nout][pitch] (cols col0..col0+Kreal, zero-fill to Kpad)
// into WMMA-B fragment order, bf16, with the ISA B lane/K map.
__global__ void pack_b_kernel(const float* __restrict__ W, __bf16* __restrict__ out,
                              int Nout, int Kpad, int Kreal, int pitch, int col0)
{
  const int KT = Kpad >> 5;
  const int total = Nout * Kpad;
  for (int idx = blockIdx.x * blockDim.x + threadIdx.x; idx < total;
       idx += gridDim.x * blockDim.x){
    int frag = idx >> 9;
    int e    = idx & 511;
    int lane = e >> 4;
    int j    = e & 15;
    int nt = frag / KT, kt = frag - nt * KT;
    int n  = (nt << 4) + (lane & 15);
    int kb = (kt << 5) + ((lane & 16) ? 8 : 0);
    int k  = kb + (j & 7) + ((j >= 8) ? 16 : 0);
    out[idx] = (k < Kreal) ? f2bf(W[(size_t)n * pitch + col0 + k]) : f2bf(0.f);
  }
}

__global__ void __launch_bounds__(TPB_, 1)
gru_plant_kernel(
    const float* __restrict__ x_cv, const float* __restrict__ x_tgt,
    const float* __restrict__ pv_init, const int* __restrict__ scenario,
    const float* __restrict__ emb,
    const float* __restrict__ bih_e0, const float* __restrict__ bhh_e0,
    const float* __restrict__ bih_e1, const float* __restrict__ bhh_e1,
    const float* __restrict__ dWih0_full,
    const float* __restrict__ bih_d0, const float* __restrict__ bhh_d0,
    const float* __restrict__ bih_d1, const float* __restrict__ bhh_d1,
    const float* __restrict__ fc1_b, const float* __restrict__ fc2_W,
    const float* __restrict__ fc2_b,
    const float* __restrict__ he1_b, const float* __restrict__ he2_W,
    const float* __restrict__ he2_b,
    const __bf16* __restrict__ Pe_ih0, const __bf16* __restrict__ Pe_hh0,
    const __bf16* __restrict__ Pe_ih1, const __bf16* __restrict__ Pe_hh1,
    const __bf16* __restrict__ Pd_ih0x, const __bf16* __restrict__ Pd_hh0,
    const __bf16* __restrict__ Pd_ih1, const __bf16* __restrict__ Pd_hh1,
    const __bf16* __restrict__ Pfc1, const __bf16* __restrict__ Phe1,
    float* __restrict__ out)
{
  extern __shared__ char smem[];
  __bf16* h1   = (__bf16*)smem;                 // 16x512 bf16  (enc h1 / dec d1)
  __bf16* h2   = h1 + 16 * H_;                  // 16x512 bf16  (enc h2 / dec d2)
  __bf16* xA   = h2 + 16 * H_;                  // 16xXPAD bf16 input tile
  float*  gsum = (float*)(xA + 16 * XPAD_);     // 16x1536 f32  fused gi+gh
  float*  hnn  = gsum + 16 * G3_;               // 16x512  f32  h-part of n-gate
  float*  pvs  = hnn + 16 * H_;                 // 16x8    f32  autoregressive pv
  int*    scen = (int*)(pvs + 16 * 8);          // 16

  const int tid  = threadIdx.x;
  const int wave = tid >> 5, lane = tid & 31;
  const int b0   = blockIdx.x * BTILE_;

  for (int i = tid; i < 16 * H_; i += TPB_){ h1[i] = f2bf(0.f); h2[i] = f2bf(0.f); }
  for (int i = tid; i < 16 * XPAD_; i += TPB_) xA[i] = f2bf(0.f);  // incl. K pad
  for (int i = tid; i < 16 * NPV_; i += TPB_)
    pvs[(i / NPV_) * 8 + (i % NPV_)] = pv_init[(b0 + i / NPV_) * NPV_ + (i % NPV_)];
  if (tid < 16) scen[tid] = scenario[b0 + tid];
  __syncthreads();

  // ------------------------------ encoder --------------------------------
  for (int t = 0; t < TE_; ++t){
    for (int i = tid; i < 16 * 160; i += TPB_){
      int m = i / 160, c = i - m * 160;
      float v = (c < NIN_) ? x_cv[((size_t)(b0 + m) * TE_ + t) * NIN_ + c]
                           : emb[scen[m] * EMB_ + (c - NIN_)];
      xA[m * XPAD_ + c] = f2bf(v);
    }
    __syncthreads();
    sched_fence();
    gates_gemm<6>(xA, XPAD_, Pe_ih0, h1, Pe_hh0, bih_e0, bhh_e0, gsum, hnn, wave, lane);
    sched_fence();
    __syncthreads();
    gru_update(gsum, hnn, h1, tid);
    __syncthreads();
    sched_fence();
    gates_gemm<16>(h1, H_, Pe_ih1, h2, Pe_hh1, bih_e1, bhh_e1, gsum, hnn, wave, lane);
    sched_fence();
    __syncthreads();
    gru_update(gsum, hnn, h2, tid);
    __syncthreads();
  }

  float* out_pv = out;
  float* out_he = out + (size_t)B_ * T_ * NPV_;

  // ------------------------------ decoder --------------------------------
  for (int t = 0; t < T_; ++t){
    for (int i = tid; i < 16 * NIN_; i += TPB_){
      int m = i >> 7, c = i & 127;
      xA[m * NIN_ + c] = f2bf(x_tgt[((size_t)(b0 + m) * T_ + t) * NIN_ + c]);
    }
    __syncthreads();
    sched_fence();
    gates_gemm<4>(xA, NIN_, Pd_ih0x, h1, Pd_hh0, bih_d0, bhh_d0, gsum, hnn, wave, lane);
    sched_fence();
    __syncthreads();
    // pv (5-dim) contribution to gi, VALU: gsum[m][n] += pv[m]·Wih0[n][128:133]
    for (int i = tid; i < 16 * G3_; i += TPB_){
      int m = i / G3_, n = i - m * G3_;
      const float* wrow = dWih0_full + (size_t)n * (NIN_ + NPV_) + NIN_;
      float s = 0.f;
#pragma unroll
      for (int k = 0; k < NPV_; ++k) s += pvs[m * 8 + k] * wrow[k];
      gsum[i] += s;
    }
    __syncthreads();
    gru_update(gsum, hnn, h1, tid);
    __syncthreads();
    sched_fence();
    gates_gemm<16>(h1, H_, Pd_ih1, h2, Pd_hh1, bih_d1, bhh_d1, gsum, hnn, wave, lane);
    sched_fence();
    __syncthreads();
    gru_update(gsum, hnn, h2, tid);
    __syncthreads();

    // heads layer 1 (ReLU(d2@W^T+b)): waves 0-7 -> fc1, waves 8-15 -> he1.
    {
      float* r1   = hnn;              // reuse: 16x128 f32
      float* r1he = hnn + 16 * 128;   // 16x128 f32
      const __bf16* W    = (wave < 8) ? Pfc1  : Phe1;
      const float* bias  = (wave < 8) ? fc1_b : he1_b;
      float*       dst   = (wave < 8) ? r1    : r1he;
      const int nt = wave & 7;
      sched_fence();
      v8f acc = {};
      acc = gemm_k<16>(h2, H_, W, nt, lane, acc);
      sched_fence();
      const int n  = (nt << 4) + (lane & 15);
      const int mb = (lane & 16) ? 8 : 0;
      const float bv = bias[n];
#pragma unroll
      for (int i = 0; i < 8; ++i)
        dst[(mb + i) * 128 + n] = fmaxf(acc[i] + bv, 0.f);
    }
    __syncthreads();

    // heads layer 2 + outputs + pv feedback (tiny: VALU dot products)
    if (tid < 16 * NPV_){
      int m = tid / NPV_, o = tid - m * NPV_;
      const float* r1 = hnn;
      float s = fc2_b[o];
      for (int k = 0; k < 128; ++k) s += r1[m * 128 + k] * fc2_W[o * 128 + k];
      out_pv[((size_t)(b0 + m) * T_ + t) * NPV_ + o] = s;
      pvs[m * 8 + o] = s;
    } else if (tid >= 128 && tid < 128 + 16 * NHE_){
      int q = tid - 128; int m = q >> 4, o = q & 15;
      const float* r1he = hnn + 16 * 128;
      float s = he2_b[o];
      for (int k = 0; k < 128; ++k) s += r1he[m * 128 + k] * he2_W[o * 128 + k];
      out_he[((size_t)(b0 + m) * T_ + t) * NHE_ + o] = s;
    }
    __syncthreads();
  }
}

extern "C" void kernel_launch(void* const* d_in, const int* in_sizes, int n_in,
                              void* d_out, int out_size, void* d_ws, size_t ws_size,
                              hipStream_t stream)
{
  (void)in_sizes; (void)n_in; (void)out_size; (void)ws_size;

  const float* x_cv   = (const float*)d_in[0];
  const float* x_tgt  = (const float*)d_in[1];
  const float* pv_init= (const float*)d_in[2];
  const int*   scen   = (const int*)  d_in[3];
  const float* emb    = (const float*)d_in[4];
  const float* eWih0  = (const float*)d_in[5];
  const float* eWhh0  = (const float*)d_in[6];
  const float* ebih0  = (const float*)d_in[7];
  const float* ebhh0  = (const float*)d_in[8];
  const float* eWih1  = (const float*)d_in[9];
  const float* eWhh1  = (const float*)d_in[10];
  const float* ebih1  = (const float*)d_in[11];
  const float* ebhh1  = (const float*)d_in[12];
  const float* dWih0  = (const float*)d_in[13];
  const float* dWhh0  = (const float*)d_in[14];
  const float* dbih0  = (const float*)d_in[15];
  const float* dbhh0  = (const float*)d_in[16];
  const float* dWih1  = (const float*)d_in[17];
  const float* dWhh1  = (const float*)d_in[18];
  const float* dbih1  = (const float*)d_in[19];
  const float* dbhh1  = (const float*)d_in[20];
  const float* fc1W   = (const float*)d_in[21];
  const float* fc1b   = (const float*)d_in[22];
  const float* fc2W   = (const float*)d_in[23];
  const float* fc2b   = (const float*)d_in[24];
  const float* he1W   = (const float*)d_in[25];
  const float* he1b   = (const float*)d_in[26];
  const float* he2W   = (const float*)d_in[27];
  const float* he2b   = (const float*)d_in[28];

  // bf16 packed-weight workspace (~10.7 MB total)
  __bf16* ws = (__bf16*)d_ws;
  size_t off = 0;
  __bf16* Pe_ih0  = ws + off; off += (size_t)96 * 6  * 512;   // K=160 pad 192
  __bf16* Pe_hh0  = ws + off; off += (size_t)96 * 16 * 512;   // K=512
  __bf16* Pe_ih1  = ws + off; off += (size_t)96 * 16 * 512;
  __bf16* Pe_hh1  = ws + off; off += (size_t)96 * 16 * 512;
  __bf16* Pd_ih0x = ws + off; off += (size_t)96 * 4  * 512;   // K=128 of 133
  __bf16* Pd_hh0  = ws + off; off += (size_t)96 * 16 * 512;
  __bf16* Pd_ih1  = ws + off; off += (size_t)96 * 16 * 512;
  __bf16* Pd_hh1  = ws + off; off += (size_t)96 * 16 * 512;
  __bf16* Pfc1    = ws + off; off += (size_t)8  * 16 * 512;   // 128x512
  __bf16* Phe1    = ws + off; off += (size_t)8  * 16 * 512;

  auto pack = [&](const float* W, __bf16* dst, int Nout, int Kpad, int Kreal,
                  int pitch, int col0){
    int total = Nout * Kpad;
    pack_b_kernel<<<(total + 255) / 256, 256, 0, stream>>>(W, dst, Nout, Kpad,
                                                           Kreal, pitch, col0);
  };
  pack(eWih0, Pe_ih0, G3_, 192, 160, 160, 0);
  pack(eWhh0, Pe_hh0, G3_, 512, 512, 512, 0);
  pack(eWih1, Pe_ih1, G3_, 512, 512, 512, 0);
  pack(eWhh1, Pe_hh1, G3_, 512, 512, 512, 0);
  pack(dWih0, Pd_ih0x, G3_, 128, 128, 133, 0);
  pack(dWhh0, Pd_hh0, G3_, 512, 512, 512, 0);
  pack(dWih1, Pd_ih1, G3_, 512, 512, 512, 0);
  pack(dWhh1, Pd_hh1, G3_, 512, 512, 512, 0);
  pack(fc1W,  Pfc1,   128, 512, 512, 512, 0);
  pack(he1W,  Phe1,   128, 512, 512, 512, 0);

  const size_t smem =
      (size_t)16 * H_ * 2 * 2     // h1, h2 (bf16)
    + (size_t)16 * XPAD_ * 2      // xA
    + (size_t)16 * G3_ * 4        // gsum
    + (size_t)16 * H_ * 4         // hnn / head scratch
    + (size_t)16 * 8 * 4          // pv
    + 64;                         // scen
  (void)hipFuncSetAttribute((const void*)gru_plant_kernel,
                            hipFuncAttributeMaxDynamicSharedMemorySize, (int)smem);

  gru_plant_kernel<<<NBLK_, TPB_, smem, stream>>>(
      x_cv, x_tgt, pv_init, scen, emb,
      ebih0, ebhh0, ebih1, ebhh1,
      dWih0, dbih0, dbhh0, dbih1, dbhh1,
      fc1b, fc2W, fc2b, he1b, he2W, he2b,
      Pe_ih0, Pe_hh0, Pe_ih1, Pe_hh1,
      Pd_ih0x, Pd_hh0, Pd_ih1, Pd_hh1,
      Pfc1, Phe1,
      (float*)d_out);
}
